// SS2D_9577777070322
// MI455X (gfx1250) — matrix operand order repaired and checked by reading, not compiled, not run on hardware
//
#include <hip/hip_runtime.h>
#include <cstdint>
#include <cstddef>

// ---------------- problem constants ----------------
#define B_   8
#define L_   1024
#define DM_  192
#define DI_  384
#define K_   8
#define N_   16
#define R_   12
#define C44  44
#define CP   48     // padded projection rows (44 -> 48)

typedef __bf16 bf16;
typedef bf16   bf16x8  __attribute__((ext_vector_type(8)));
typedef bf16   bf16x16 __attribute__((ext_vector_type(16)));
typedef float  f32x8   __attribute__((ext_vector_type(8)));
typedef unsigned int u32x4 __attribute__((ext_vector_type(4)));
typedef int    i32x4  __attribute__((ext_vector_type(4)));
typedef int    i32x8  __attribute__((ext_vector_type(8)));

// ---------------- device helpers ----------------
__device__ __forceinline__ bf16 tobf(float f) {
  union { float f; uint32_t u; } v; v.f = f;
  uint32_t r = v.u + 0x7FFFu + ((v.u >> 16) & 1u);   // round-to-nearest-even
  union { uint16_t s; bf16 b; } o; o.s = (uint16_t)(r >> 16);
  return o.b;
}

// spatial position (h*32+w) touched by scan direction `dir` at scan step l
__device__ __forceinline__ int pos_of(int dir, int l) {
  int row = l >> 5, c = l & 31;
  int cs = (row & 1) ? (31 - c) : c;          // snake column
  if (dir == 0) return row * 32 + cs;         // enc(xc)
  if (dir == 1) return cs * 32 + row;         // enc(xc^T)
  if (dir == 2) return cs * 32 + (31 - row);  // enc(rot90(xc))
  return row * 32 + (31 - cs);                // enc(rot90(xc)^T)
}

// A frag (16x32 bf16, ISA 7.12.2): lane half kh holds K = kh*8..kh*8+7 and 16+kh*8..
__device__ __forceinline__ bf16x16 ld_frag_a(const bf16* rowp, int k0, int kh) {
  bf16x8 lo = *reinterpret_cast<const bf16x8*>(rowp + k0 + kh * 8);
  bf16x8 hi = *reinterpret_cast<const bf16x8*>(rowp + k0 + 16 + kh * 8);
  bf16x16 a;
#pragma unroll
  for (int j = 0; j < 8; ++j) { a[j] = lo[j]; a[8 + j] = hi[j]; }
  return a;
}

// B frag (32x16 bf16): lane group kh holds K = kh*8..kh*8+7 and 16+kh*8.. (same as A)
__device__ __forceinline__ bf16x16 ld_frag_b(const bf16* colp, int k0, int kh) {
  bf16x8 lo = *reinterpret_cast<const bf16x8*>(colp + k0 + kh * 8);
  bf16x8 hi = *reinterpret_cast<const bf16x8*>(colp + k0 + 16 + kh * 8);
  bf16x16 b;
#pragma unroll
  for (int j = 0; j < 8; ++j) { b[j] = lo[j]; b[8 + j] = hi[j]; }
  return b;
}

// Tensor Data Mover: async 2D tile load (44 rows x 16 f32, row stride L_) into LDS.
// D# per CDNA5 ISA §8.3/8.4: group0 {count=1 | lds_addr | global_addr | type=2},
// group1 {data_size=4B, tensor=tile=16x44, dim0_stride=L_}. Tracked by TENSORcnt.
__device__ __forceinline__ void tdm_load_tile(uint32_t lds_addr, uint64_t gaddr) {
  u32x4 g0;
  g0[0] = 1u;                                             // count=1 (valid user D#)
  g0[1] = lds_addr;                                       // LDS byte address
  g0[2] = (uint32_t)(gaddr & 0xFFFFFFFFu);                // global_addr[31:0]
  g0[3] = (uint32_t)((gaddr >> 32) & 0x01FFFFFFu)         // global_addr[56:32]
        | (2u << 30);                                     // type=2 ("image")
  i32x8 g1;
  g1[0] = (int)(2u << 16);        // workgroup_mask=0, data_size=2 (4 bytes)
  g1[1] = (int)(16u << 16);       // tensor_dim0 = 16      (bits 63:48)
  g1[2] = (int)(44u << 16);       // tensor_dim1 = 44      (bits 95:80)
  g1[3] = (int)(16u << 16);       // tile_dim0   = 16      (bits 127:112)
  g1[4] = 44;                     // tile_dim1   = 44
  g1[5] = L_;                     // tensor_dim0_stride lo = 1024 elements
  g1[6] = 0;                      // stride hi / dim1 stride (unused, 2D tile)
  g1[7] = 0;
  i32x4 z4 = {0, 0, 0, 0};
#if __clang_major__ >= 23
  i32x8 z8 = {0, 0, 0, 0, 0, 0, 0, 0};
  __builtin_amdgcn_tensor_load_to_lds(g0, g1, z4, z4, z8, 0);
#else
  __builtin_amdgcn_tensor_load_to_lds(g0, g1, z4, z4, 0);
#endif
}

// ---------------- kernels ----------------

// generic f32 -> bf16 conversion
__global__ void cvt_bf16_kernel(const float* __restrict__ in, bf16* __restrict__ out, int n) {
  int i = blockIdx.x * 256 + threadIdx.x;
  if (i < n) out[i] = tobf(in[i]);
}

// GEMM1: xz = x(8192x192) @ W_in^T(192x768); split columns into xc (B,L,DI) and z (B,L,DI)
__global__ void gemm_in_kernel(const bf16* __restrict__ xb, const bf16* __restrict__ wb,
                               float* __restrict__ xc, float* __restrict__ z) {
  int wave = threadIdx.x >> 5, lane = threadIdx.x & 31;
  int t  = blockIdx.x * 8 + wave;        // 24576 tiles: 512 x 48
  int mt = t / 48, nt = t % 48;
  int m  = lane & 15, kh = lane >> 4;
  const bf16* Ab = xb + (size_t)(mt * 16 + m) * DM_;
  const bf16* Bb = wb + (size_t)(nt * 16 + m) * DM_;   // W_in row = output col, contiguous K
  f32x8 acc = {};
#pragma unroll
  for (int kt = 0; kt < 6; ++kt) {                     // K = 192
    int k0 = kt * 32;
    bf16x16 a = ld_frag_a(Ab, k0, kh);
    bf16x16 b = ld_frag_b(Bb, k0, kh);
    acc = __builtin_amdgcn_wmma_f32_16x16x32_bf16(false, a, false, b, (short)0, acc, false, false);
  }
  int col = nt * 16 + m;
#pragma unroll
  for (int vj = 0; vj < 8; ++vj) {                     // D: lanes 0-15 M=vj, lanes 16-31 M=8+vj
    int g = mt * 16 + kh * 8 + vj;                     // g = b*L + l
    int bb = g >> 10, l = g & (L_ - 1);
    float val = acc[vj];
    if (col < DI_) xc[((size_t)(bb * L_ + l)) * DI_ + col] = val;
    else           z [((size_t)(bb * L_ + l)) * DI_ + (col - DI_)] = val;
  }
}

// depthwise 3x3 conv (SAME) + bias + SiLU, (B,L,DI) layout; writes f32 + bf16 copies
__global__ void conv_silu_kernel(const float* __restrict__ xc, const float* __restrict__ cw,
                                 const float* __restrict__ cb, float* __restrict__ xcv,
                                 bf16* __restrict__ xcvb) {
  int idx = blockIdx.x * 256 + threadIdx.x;            // (b*L + l)*DI + d, exact multiple
  int d  = idx % DI_;
  int bl = idx / DI_;
  int l  = bl & (L_ - 1), b = bl >> 10;
  int h  = l >> 5, w = l & 31;
  float acc = cb[d];
#pragma unroll
  for (int i = 0; i < 3; ++i) {
    int hh = h + i - 1;
    if (hh < 0 || hh >= 32) continue;
#pragma unroll
    for (int j = 0; j < 3; ++j) {
      int ww = w + j - 1;
      if (ww < 0 || ww >= 32) continue;
      acc = fmaf(cw[d * 9 + i * 3 + j], xc[((size_t)(b * L_ + hh * 32 + ww)) * DI_ + d], acc);
    }
  }
  float s = acc / (1.f + __expf(-acc));                // SiLU
  xcv[idx]  = s;
  xcvb[idx] = tobf(s);
}

// GEMM2: P[b,k,c,l_scan] = sum_d x_proj_w[k,c,d] * xconv[b,d,pos(k&3,l)]  (44x1024x384 per b,k)
__global__ void gemm_proj_kernel(const bf16* __restrict__ xpwb, const bf16* __restrict__ xcvb,
                                 float* __restrict__ P) {
  int wave = threadIdx.x >> 5, lane = threadIdx.x & 31;
  int t  = blockIdx.x * 8 + wave;                      // 12288 tiles = 8b * 8k * 3ct * 64lt
  int lt = t & 63; int r = t >> 6;
  int ct = r % 3;  r /= 3;
  int k  = r & 7;  int b = r >> 3;
  int m  = lane & 15, kh = lane >> 4;
  int c  = ct * 16 + m;
  int ca = (c < C44) ? c : 0;                          // clamp: padded rows produce unused garbage
  const bf16* Ab = xpwb + ((size_t)(k * C44 + ca)) * DI_;
  int pos = pos_of(k & 3, lt * 16 + m);
  const bf16* Bb = xcvb + ((size_t)(b * L_ + pos)) * DI_;
  f32x8 acc = {};
#pragma unroll
  for (int kt = 0; kt < 12; ++kt) {                    // K = 384
    int k0 = kt * 32;
    bf16x16 a  = ld_frag_a(Ab, k0, kh);
    bf16x16 bb = ld_frag_b(Bb, k0, kh);
    acc = __builtin_amdgcn_wmma_f32_16x16x32_bf16(false, a, false, bb, (short)0, acc, false, false);
  }
#pragma unroll
  for (int vj = 0; vj < 8; ++vj) {
    int crow = ct * 16 + kh * 8 + vj;                  // 0..47 (padded stride CP)
    P[((size_t)((b * K_ + k) * CP + crow)) * L_ + lt * 16 + m] = acc[vj];
  }
}

// selective scan: one block (384 thr = 12 waves) per (b, dir). Fwd pass (k=dir) stores,
// bwd pass (k=dir+4, reversed order) accumulates into the same spatial buffer.
// Per-tile operands (44 x 16 f32) are staged into LDS by the Tensor Data Mover,
// double-buffered: wave 0 issues tile T+1 while all waves compute tile T.
__global__ void scan_kernel(const float* __restrict__ P, const float* __restrict__ u_sp,
                            const float* __restrict__ dtw, const float* __restrict__ dtb,
                            const float* __restrict__ Alogs, const float* __restrict__ Ds,
                            float* __restrict__ S) {
  int blk = blockIdx.x;                 // b*4 + dir
  int b = blk >> 2, dir = blk & 3;
  int di = threadIdx.x;                 // 0..383
  __shared__ float sP[2][C44 * 16];

  for (int pass = 0; pass < 2; ++pass) {
    int k = dir + pass * 4;
    float w[R_];
#pragma unroll
    for (int j = 0; j < R_; ++j) w[j] = dtw[((size_t)(k * DI_ + di)) * R_ + j];
    float bias = dtb[k * DI_ + di];
    float Dv   = Ds [k * DI_ + di];
    float Av[N_];
#pragma unroll
    for (int n = 0; n < N_; ++n) Av[n] = -__expf(Alogs[((size_t)(k * DI_ + di)) * N_ + n]);
    float h[N_];
#pragma unroll
    for (int n = 0; n < N_; ++n) h[n] = 0.f;

    const float* Pk = P + ((size_t)(b * K_ + k)) * CP * L_;
    uint64_t pbase = (uint64_t)(uintptr_t)Pk;

    if (threadIdx.x < 32) {                          // wave 0: prime first tile
      int Ti0 = pass ? 63 : 0;
      tdm_load_tile((uint32_t)(uintptr_t)&sP[0][0], pbase + (uint64_t)Ti0 * 64u);
    }
    for (int T = 0; T < 64; ++T) {
      int Ti = pass ? (63 - T) : T;
      if (threadIdx.x < 32) {                        // wave 0: prefetch next tile via TDM
        if (T + 1 < 64) {
          int Tn = pass ? (63 - (T + 1)) : (T + 1);
          tdm_load_tile((uint32_t)(uintptr_t)&sP[(T + 1) & 1][0], pbase + (uint64_t)Tn * 64u);
          __builtin_amdgcn_s_wait_tensorcnt((short)1);   // current tile complete
        } else {
          __builtin_amdgcn_s_wait_tensorcnt((short)0);
        }
      }
      __syncthreads();                               // tile (T&1) visible to all waves
      const float* sp = sP[T & 1];
#pragma unroll 1
      for (int s = 0; s < 16; ++s) {
        int tt  = pass ? (15 - s) : s;
        int l   = Ti * 16 + tt;
        int pos = pos_of(dir, l);
        float uv = u_sp[((size_t)(b * L_ + pos)) * DI_ + di];
        float x = bias;
#pragma unroll
        for (int j = 0; j < R_; ++j) x = fmaf(w[j], sp[j * 16 + tt], x);
        float delta = (x > 20.f) ? x : log1pf(__expf(x));   // softplus
        float du = delta * uv;
        float y = 0.f;
#pragma unroll
        for (int n = 0; n < N_; ++n) {
          float dA = __expf(delta * Av[n]);
          h[n] = fmaf(dA, h[n], du * sp[(R_ + n) * 16 + tt]);        // B_n
          y    = fmaf(h[n], sp[(R_ + N_ + n) * 16 + tt], y);         // C_n
        }
        y = fmaf(Dv, uv, y);
        size_t so = ((size_t)((b * 4 + dir) * L_ + pos)) * DI_ + di;
        if (pass == 0) S[so] = y; else S[so] += y;   // same thread wrote it in pass 0
      }
      __syncthreads();                               // protect buffer before TDM overwrite
    }
  }
}

// sum 4 direction buffers + LayerNorm + SiLU(z) gate -> bf16 for output GEMM
__global__ void ln_gate_kernel(const float* __restrict__ S, const float* __restrict__ z,
                               const float* __restrict__ lnw, const float* __restrict__ lnb,
                               bf16* __restrict__ yg) {
  int bl = blockIdx.x;                  // b*L + l
  int b = bl >> 10, l = bl & (L_ - 1);
  int tid = threadIdx.x;
  __shared__ float red[128];
  float v[3];
  float sum = 0.f;
#pragma unroll
  for (int j = 0; j < 3; ++j) {
    int di = j * 128 + tid;
    float acc = 0.f;
#pragma unroll
    for (int d = 0; d < 4; ++d)
      acc += S[((size_t)((b * 4 + d) * L_ + l)) * DI_ + di];
    v[j] = acc; sum += acc;
  }
  red[tid] = sum; __syncthreads();
  for (int s = 64; s > 0; s >>= 1) { if (tid < s) red[tid] += red[tid + s]; __syncthreads(); }
  float mu = red[0] * (1.f / DI_);
  __syncthreads();
  float sq = 0.f;
#pragma unroll
  for (int j = 0; j < 3; ++j) { float d0 = v[j] - mu; sq += d0 * d0; }
  red[tid] = sq; __syncthreads();
  for (int s = 64; s > 0; s >>= 1) { if (tid < s) red[tid] += red[tid + s]; __syncthreads(); }
  float rstd = rsqrtf(red[0] * (1.f / DI_) + 1e-5f);
#pragma unroll
  for (int j = 0; j < 3; ++j) {
    int di = j * 128 + tid;
    float zv = z[((size_t)bl) * DI_ + di];
    float g  = zv / (1.f + __expf(-zv));                 // SiLU
    float o  = (v[j] - mu) * rstd * lnw[di] + lnb[di];
    yg[((size_t)bl) * DI_ + di] = tobf(o * g);
  }
}

// GEMM4: out = yg(8192x384) @ W_out^T(384x192)
__global__ void gemm_out_kernel(const bf16* __restrict__ yg, const bf16* __restrict__ wob,
                                float* __restrict__ out) {
  int wave = threadIdx.x >> 5, lane = threadIdx.x & 31;
  int t  = blockIdx.x * 8 + wave;       // 6144 tiles: 512 x 12
  int mt = t / 12, nt = t % 12;
  int m  = lane & 15, kh = lane >> 4;
  const bf16* Ab = yg  + (size_t)(mt * 16 + m) * DI_;
  const bf16* Bb = wob + (size_t)(nt * 16 + m) * DI_;
  f32x8 acc = {};
#pragma unroll
  for (int kt = 0; kt < 12; ++kt) {     // K = 384
    int k0 = kt * 32;
    bf16x16 a = ld_frag_a(Ab, k0, kh);
    bf16x16 b = ld_frag_b(Bb, k0, kh);
    acc = __builtin_amdgcn_wmma_f32_16x16x32_bf16(false, a, false, b, (short)0, acc, false, false);
  }
  int col = nt * 16 + m;
#pragma unroll
  for (int vj = 0; vj < 8; ++vj) {
    int g = mt * 16 + kh * 8 + vj;
    out[(size_t)g * DM_ + col] = acc[vj];
  }
}

// ---------------- host launcher ----------------
extern "C" void kernel_launch(void* const* d_in, const int* in_sizes, int n_in,
                              void* d_out, int out_size, void* d_ws, size_t ws_size,
                              hipStream_t stream) {
  const float* x     = (const float*)d_in[0];
  const float* W_in  = (const float*)d_in[1];
  const float* convw = (const float*)d_in[2];
  const float* convb = (const float*)d_in[3];
  const float* xpw   = (const float*)d_in[4];
  const float* dtw   = (const float*)d_in[5];
  const float* dtb   = (const float*)d_in[6];
  const float* Alogs = (const float*)d_in[7];
  const float* Ds    = (const float*)d_in[8];
  const float* lnw   = (const float*)d_in[9];
  const float* lnb   = (const float*)d_in[10];
  const float* W_out = (const float*)d_in[11];

  char* ws = (char*)d_ws;
  size_t off = 0;
  auto take = [&](size_t bytes) -> char* {
    char* p = ws + off; off += (bytes + 255) & ~(size_t)255; return p;
  };

  const size_t nX   = (size_t)B_ * L_ * DM_;     // 1572864
  const size_t nWin = (size_t)2 * DI_ * DM_;     // 147456
  const size_t nXPW = (size_t)K_ * C44 * DI_;    // 135168
  const size_t nWo  = (size_t)DM_ * DI_;         // 73728
  const size_t nBLD = (size_t)B_ * L_ * DI_;     // 3145728

  bf16*  xb    = (bf16*) take(nX   * 2);
  bf16*  winb  = (bf16*) take(nWin * 2);
  float* xc    = (float*)take(nBLD * 4);                       // pre-conv (B,L,DI)
  float* zbuf  = (float*)take(nBLD * 4);                       // gate branch
  float* xcv   = (float*)take(nBLD * 4);                       // conv out f32
  bf16*  xcvb  = (bf16*) take(nBLD * 2);                       // conv out bf16
  bf16*  xpwb  = (bf16*) take(nXPW * 2);
  float* P     = (float*)take((size_t)B_ * K_ * CP * L_ * 4);  // scan-ordered projections
  float* S     = (float*)take((size_t)B_ * 4 * L_ * DI_ * 4);  // per-direction scan sums
  bf16*  yg    = (bf16*) take(nBLD * 2);
  bf16*  woutb = (bf16*) take(nWo  * 2);

  // bf16 conversions for WMMA operands
  cvt_bf16_kernel<<<(int)((nX   + 255) / 256), 256, 0, stream>>>(x,     xb,    (int)nX);
  cvt_bf16_kernel<<<(int)((nWin + 255) / 256), 256, 0, stream>>>(W_in,  winb,  (int)nWin);
  cvt_bf16_kernel<<<(int)((nXPW + 255) / 256), 256, 0, stream>>>(xpw,   xpwb,  (int)nXPW);
  cvt_bf16_kernel<<<(int)((nWo  + 255) / 256), 256, 0, stream>>>(W_out, woutb, (int)nWo);

  // input projection (WMMA)
  gemm_in_kernel<<<3072, 256, 0, stream>>>(xb, winb, xc, zbuf);

  // depthwise conv + SiLU
  conv_silu_kernel<<<(int)(nBLD / 256), 256, 0, stream>>>(xc, convw, convb, xcv, xcvb);

  // x_proj einsum in scan order (WMMA)
  gemm_proj_kernel<<<1536, 256, 0, stream>>>(xpwb, xcvb, P);

  // selective scan: fwd + bwd per direction, TDM-staged operands
  scan_kernel<<<B_ * 4, DI_, 0, stream>>>(P, xcv, dtw, dtb, Alogs, Ds, S);

  // direction-sum + LayerNorm + gate
  ln_gate_kernel<<<B_ * L_, 128, 0, stream>>>(S, zbuf, lnw, lnb, yg);

  // output projection (WMMA)
  gemm_out_kernel<<<768, 256, 0, stream>>>(yg, woutb, (float*)d_out);
}
